// NT_Xent_88029649699158
// MI455X (gfx1250) — matrix operand order repaired and checked by reading, not compiled, not run on hardware
//
#include <hip/hip_runtime.h>

// ---------------------------------------------------------------------------
// NT-Xent (SimCLR) fused loss for MI455X / gfx1250.
//   B = 4096, N = 8192, D = 128, T = 0.5
// Strategy:
//   prep:   row-normalize z (folding the cosine denominator), split into
//           bf16 hi + bf16 lo so that hi*hi + hi*lo + lo*hi ~ fp32 dot.
//   main:   fused tiled GEMM (v_wmma_f32_16x16x32_bf16) + streaming
//           fixed-max exp-sum (|logits| <= 1/T = 2, so max is the constant 2).
//           No per-element masking in the hot loop:
//             - diagonal term == exp2(0) == 1, subtracted analytically later
//             - positive pair lives in ONE wave-uniform column tile (r0 +- B),
//               tested with a scalar (readfirstlane) compare -> s_cbranch.
//           Partials merged with global atomicAdd (fixed max => plain sums).
//   reduce: loss = mean( ln(S_i - 1) + 2 - pos_i ).
// ---------------------------------------------------------------------------

typedef __attribute__((ext_vector_type(16))) __bf16 v16bf;
typedef __attribute__((ext_vector_type(8)))  float  v8f;

#define NT_B 4096
#define NT_N 8192
#define NT_D 128

// (1/T) * log2(e)  with T = 0.5  -> 2 * 1.4426950408889634
#define SCALE_LOG2E 2.8853900817779268f
#define LN2F        0.6931471805599453f

__device__ __forceinline__ unsigned short f2bf_rne(float f) {
    unsigned u = __float_as_uint(f);
    u += 0x7FFFu + ((u >> 16) & 1u);
    return (unsigned short)(u >> 16);
}
__device__ __forceinline__ float bf2f(unsigned short h) {
    return __uint_as_float(((unsigned)h) << 16);
}

// ---------------------------------------------------------------------------
// Kernel 1: normalize rows, emit bf16 hi/lo split, zero the accumulators.
// block = 128 threads = 4 waves, one row per wave. grid = N/4 = 2048.
// ---------------------------------------------------------------------------
__global__ __launch_bounds__(128) void ntxent_prep(
    const float* __restrict__ zi, const float* __restrict__ zj,
    unsigned short* __restrict__ zhi, unsigned short* __restrict__ zlo,
    float* __restrict__ S, float* __restrict__ P)
{
    const int lane = threadIdx.x & 31;
    const int wave = threadIdx.x >> 5;
    const int row  = blockIdx.x * 4 + wave;

    const float* src = (row < NT_B) ? (zi + (size_t)row * NT_D)
                                    : (zj + (size_t)(row - NT_B) * NT_D);
    float x[4];
    float ss = 0.0f;
#pragma unroll
    for (int c = 0; c < 4; ++c) {
        x[c] = src[lane + 32 * c];
        ss += x[c] * x[c];
    }
#pragma unroll
    for (int m = 16; m >= 1; m >>= 1) ss += __shfl_xor(ss, m);

    // 1/max(norm, sqrt(eps)) reproduces max(|zi||zj|, eps) for these scales.
    const float inv = 1.0f / fmaxf(sqrtf(ss), 1e-4f);

#pragma unroll
    for (int c = 0; c < 4; ++c) {
        const float f  = x[c] * inv;
        const unsigned short hi = f2bf_rne(f);
        const unsigned short lo = f2bf_rne(f - bf2f(hi));
        const size_t idx = (size_t)row * NT_D + lane + 32 * c;
        zhi[idx] = hi;
        zlo[idx] = lo;
    }
    if (lane == 0) { S[row] = 0.0f; P[row] = 0.0f; }
}

// ---------------------------------------------------------------------------
// Kernel 2: fused WMMA Gram-tile + streaming exp-sum.
// block = 256 threads = 8 waves. Each wave owns 16 rows; the block sweeps a
// 1024-column chunk (64 tiles of 16). grid = (8192/128) * (8192/1024) = 512.
// ---------------------------------------------------------------------------
union V16U { v16bf v; uint4 q[2]; };

__global__ __launch_bounds__(256) void ntxent_simlse(
    const unsigned short* __restrict__ zhi,
    const unsigned short* __restrict__ zlo,
    float* __restrict__ S, float* __restrict__ P)
{
    const int lane = threadIdx.x & 31;
    const int wave = threadIdx.x >> 5;
    const int rowGroup = blockIdx.x & 63;   // 64 groups of 128 rows
    const int colChunk = blockIdx.x >> 6;   // 8 chunks of 1024 cols

    const int r0     = rowGroup * 128 + wave * 16;      // wave's 16 rows
    const int c0base = colChunk * 1024;
    const int half   = lane >> 4;                        // 0 or 1
    const int l15    = lane & 15;

    // ---- A operand (16x32 bf16 per K-step), resident in registers.
    // lane<16: row r0+lane, K = kb+0..7 and kb+16..23
    // lane>=16: row r0+lane-16, K = kb+8..15 and kb+24..31
    const int rowA = r0 + l15;
    const int aoff = half * 8;
    V16U ahi[4], alo[4];
#pragma unroll
    for (int ks = 0; ks < 4; ++ks) {
        const int kb = ks * 32;
        const size_t base = (size_t)rowA * NT_D + kb + aoff;
        ahi[ks].q[0] = *(const uint4*)(zhi + base);
        ahi[ks].q[1] = *(const uint4*)(zhi + base + 16);
        alo[ks].q[0] = *(const uint4*)(zlo + base);
        alo[ks].q[1] = *(const uint4*)(zlo + base + 16);
    }

    float s[8];
#pragma unroll
    for (int k = 0; k < 8; ++k) s[k] = 0.0f;
    float ppv = 0.0f;   // this lane's positive-pair logit (at most one)

    const int grBase = r0 + half * 8;   // row of acc element k is grBase + k
    // The column tile holding ALL of this wave's positive pairs. It is
    // wave-uniform; readfirstlane makes that visible to the compiler so the
    // per-tile test becomes an SALU compare + s_cbranch (no EXEC traffic).
    const int posC0 =
        __builtin_amdgcn_readfirstlane(r0 + ((r0 < NT_B) ? NT_B : -NT_B));

    for (int ct = 0; ct < 64; ++ct) {
        const int c0 = c0base + ct * 16;
        // ---- B operand (32x16 bf16): lane<16 -> row c0+lane, K kb+0..7 and kb+16..23;
        //      lane>=16 -> row c0+lane-16, K kb+8..15 and kb+24..31.
        const size_t bbase = (size_t)(c0 + l15) * NT_D + half * 8;

        v8f accA = {};   // hi*hi
        v8f accB = {};   // hi*lo + lo*hi
#pragma unroll
        for (int ks = 0; ks < 4; ++ks) {
            const int kb = ks * 32;
            V16U bhi, blo;
            bhi.q[0] = *(const uint4*)(zhi + bbase + kb);
            bhi.q[1] = *(const uint4*)(zhi + bbase + kb + 16);
            blo.q[0] = *(const uint4*)(zlo + bbase + kb);
            blo.q[1] = *(const uint4*)(zlo + bbase + kb + 16);
            accA = __builtin_amdgcn_wmma_f32_16x16x32_bf16(
                false, ahi[ks].v, false, bhi.v, (short)0, accA, false, false);
            accB = __builtin_amdgcn_wmma_f32_16x16x32_bf16(
                false, ahi[ks].v, false, blo.v, (short)0, accB, false, false);
            accB = __builtin_amdgcn_wmma_f32_16x16x32_bf16(
                false, alo[ks].v, false, bhi.v, (short)0, accB, false, false);
        }

        // ---- hot epilogue: no masks, no compares.
        // exp2((a-1) * (1/T)*log2e) == exp(sim/T - 2); diagonal contributes
        // exactly 1 (||z||=1) and is subtracted analytically in the reduction.
#pragma unroll
        for (int k = 0; k < 8; ++k) {
            const float a = accA[k] + accB[k];
            s[k] += __builtin_amdgcn_exp2f(a * SCALE_LOG2E - SCALE_LOG2E);
        }

        // ---- positive-pair extraction: scalar branch, taken in 1 of 64 tiles.
        if (c0 == posC0) {
#pragma unroll
            for (int k = 0; k < 8; ++k) {
                if (l15 == k + 8 * half)
                    ppv = (accA[k] + accB[k]) * 2.0f;   // sim/T
            }
        }
    }

    // ---- reduce S across the 16 lanes of each half-wave, merge atomically.
#pragma unroll
    for (int k = 0; k < 8; ++k) {
        float v = s[k];
#pragma unroll
        for (int m = 8; m >= 1; m >>= 1) v += __shfl_xor(v, m);
        if (l15 == 0) atomicAdd(&S[grBase + k], v);
    }
    // ---- positive pair: lane (l15, half) with l15-8*half in [0,8) holds
    // row r0+l15's positive. Only the block whose chunk contains posC0 adds.
    if (posC0 >= c0base && posC0 < c0base + 1024) {   // scalar condition
        if ((l15 >> 3) == half)           // lanes that can hold a positive
            atomicAdd(&P[r0 + l15], ppv);
    }
}

// ---------------------------------------------------------------------------
// Kernel 3: loss = mean( ln(S_i - 1) + 2 - pos_i ). Single block.
// (S includes the diagonal term, which is exactly exp2(0) = 1.)
// ---------------------------------------------------------------------------
__global__ __launch_bounds__(256) void ntxent_reduce(
    const float* __restrict__ S, const float* __restrict__ P,
    float* __restrict__ out)
{
    __shared__ float red[256];
    const int tid = threadIdx.x;
    float local = 0.0f;
    for (int i = tid; i < NT_N; i += 256) {
        // v_log_f32 is log2 -> ln(x) = log2(x)*ln2 ; lse = ln(S-1) + 2
        local += __builtin_amdgcn_logf(S[i] - 1.0f) * LN2F + 2.0f - P[i];
    }
    red[tid] = local;
    __syncthreads();
    for (int off = 128; off >= 1; off >>= 1) {
        if (tid < off) red[tid] += red[tid + off];
        __syncthreads();
    }
    if (tid == 0) out[0] = red[0] / (float)NT_N;
}

// ---------------------------------------------------------------------------
extern "C" void kernel_launch(void* const* d_in, const int* in_sizes, int n_in,
                              void* d_out, int out_size, void* d_ws, size_t ws_size,
                              hipStream_t stream) {
    (void)in_sizes; (void)n_in; (void)out_size; (void)ws_size;
    const float* zi = (const float*)d_in[0];
    const float* zj = (const float*)d_in[1];
    float* out = (float*)d_out;

    // Workspace layout: zhi (2MB) | zlo (2MB) | S (32KB) | P (32KB)
    unsigned short* zhi = (unsigned short*)d_ws;
    unsigned short* zlo = zhi + (size_t)NT_N * NT_D;
    float* S = (float*)((char*)d_ws + 2ull * (size_t)NT_N * NT_D * sizeof(unsigned short));
    float* P = S + NT_N;

    ntxent_prep<<<NT_N / 4, 128, 0, stream>>>(zi, zj, zhi, zlo, S, P);
    ntxent_simlse<<<(NT_N / 128) * (NT_N / 1024), 256, 0, stream>>>(zhi, zlo, S, P);
    ntxent_reduce<<<1, 256, 0, stream>>>(S, P, out);
}
